// PixelContrastiveLearning_35510789603630
// MI455X (gfx1250) — matrix operand order repaired and checked by reading, not compiled, not run on hardware
//
#include <hip/hip_runtime.h>
#include <hip/hip_bf16.h>

#define HW     4096
#define CIN    2048
#define BATCH  8
#define MAXS   256
#define NPROJ  256
#define NATT   64
#define NTOT   320   // 256 proj + 64 attn fused in GEMM1

typedef __attribute__((ext_vector_type(16))) __bf16 v16bf;
typedef __attribute__((ext_vector_type(8)))  float  v8f;

union Frag {
    v16bf v;
    unsigned short e[16];
    uint4 q[2];
};

__device__ inline unsigned short f2bf(float f) {
    unsigned u = __float_as_uint(f);
    return (unsigned short)((u + 0x7FFFu + ((u >> 16) & 1u)) >> 16);
}
__device__ inline float bf2f(unsigned short h) {
    return __uint_as_float(((unsigned)h) << 16);
}
__device__ inline v8f wmma_bf16(const Frag& a, const Frag& b, v8f c) {
    return __builtin_amdgcn_wmma_f32_16x16x32_bf16(
        false, a.v, false, b.v, (short)0, c, false, false);
}

// ---------------------------------------------------------------- K0: select
__global__ void k_select(const int* __restrict__ labels,
                         int* sel_idx, int* sel_cnt,
                         float* sumw, float* numacc, int* hasflag) {
    int b = blockIdx.x;
    if (threadIdx.x != 0) return;
    const int* lb = labels + b * HW;
    int cnt = 0;
    for (int p = 0; p < HW; ++p) {
        if (lb[p] == 1 && cnt < MAXS) sel_idx[b * MAXS + cnt++] = p;
    }
    for (int s = cnt; s < MAXS; ++s) sel_idx[b * MAXS + s] = 0;  // padded (masked later)
    sel_cnt[b] = cnt;
    sumw[b] = 0.f; numacc[b] = 0.f; hasflag[b] = 0;
}

// ------------------------------------------- K0b: one-time weight bf16 cast
// Wb1 = [W1 rows | Wa1 rows] (320 x 2048), Wb2 = W2 (256 x 256)
__global__ __launch_bounds__(256) void k_cvtw(
    const float* __restrict__ W1, const float* __restrict__ Wa1,
    const float* __restrict__ W2,
    unsigned short* __restrict__ Wb1, unsigned short* __restrict__ Wb2) {
    int i = blockIdx.x * 256 + threadIdx.x;
    const int n1 = NTOT * CIN;               // 655360
    const int n2 = NPROJ * NPROJ;            // 65536
    if (i < n1) {
        float v = (i < NPROJ * CIN) ? W1[i] : Wa1[i - NPROJ * CIN];
        Wb1[i] = f2bf(v);
    } else if (i < n1 + n2) {
        Wb2[i - n1] = f2bf(W2[i - n1]);
    }
}

// ---------------------------------------------- K2: fused proj + attn (WMMA)
// grid = BATCH*8 blocks of 32 selected rows; 256 threads = 8 waves (2 Mw x 4 Nw)
__global__ __launch_bounds__(256) void k_proj(
    const float* __restrict__ feats,
    const unsigned short* __restrict__ Wb1, const float* __restrict__ b1,
    const unsigned short* __restrict__ Wb2, const float* __restrict__ b2,
    const float* __restrict__ ba1,
    const float* __restrict__ Wa2, const float* __restrict__ ba2,
    const int* __restrict__ sel_idx, const int* __restrict__ sel_cnt,
    float* wsel, float* sumw, unsigned short* fsn) {

    __shared__ __align__(16) unsigned short Als[2][32 * 32];  // ping-pong A tile bf16
    __shared__ __align__(16) unsigned short h1ls[32 * NTOT];  // hidden, bf16
    __shared__ int   pixs[32];
    __shared__ int   cnts;
    __shared__ float normLDS[32];

    const int bx = blockIdx.x;
    const int b  = bx >> 3;
    const int s0 = (bx & 7) * 32;
    const int tid  = threadIdx.x;
    const int wave = tid >> 5, lane = tid & 31;
    const int mw = wave & 1, nw = wave >> 1;      // 2 x 4 wave grid
    const int lr = lane & 15, kh = lane >> 4;

    if (tid < 32) pixs[tid] = sel_idx[b * MAXS + s0 + tid];
    if (tid == 0) cnts = sel_cnt[b];
    __syncthreads();

    // staging decomposition: thread -> (row m, 4 consecutive k), hoisted base ptr
    const int sm = tid >> 3;          // 0..31
    const int sk = (tid & 7) * 4;     // 0,4,..,28
    const int mypix = pixs[sm];
    const float* fbase =
        feats + (size_t)b * CIN * HW + (size_t)sk * HW + mypix;
    unsigned short* stgdst0 = &Als[0][sm * 32 + sk];
    unsigned short* stgdst1 = &Als[1][sm * 32 + sk];

    auto stage = [&](int kb, int buf) {
        const float* fp = fbase + (size_t)kb * HW;
        unsigned p0 = (unsigned)f2bf(fp[0])      | ((unsigned)f2bf(fp[HW])     << 16);
        unsigned p1 = (unsigned)f2bf(fp[2 * HW]) | ((unsigned)f2bf(fp[3 * HW]) << 16);
        *(uint2*)(buf ? stgdst1 : stgdst0) = (uint2){p0, p1};
    };

    // ---- GEMM1: x(32 x 2048) @ Wb1^T -> (32 x 320), bf16 WMMA, double-buffered
    v8f acc[5] = {};
    stage(0, 0);
    __syncthreads();
    int cur = 0;
    for (int kb = 0; kb < CIN; kb += 32) {
        if (kb + 32 < CIN) stage(kb + 32, cur ^ 1);   // overlap gather with WMMA
        Frag af;
        {   // A: lanes 0-15 hold K {0..7,16..23}, lanes 16-31 hold K {8..15,24..31}
            const unsigned short* ar = &Als[cur][(mw * 16 + lr) * 32 + kh * 8];
            af.q[0] = *(const uint4*)ar;
            af.q[1] = *(const uint4*)(ar + 16);
        }
        #pragma unroll
        for (int nt = 0; nt < 5; ++nt) {
            int col = (nw * 5 + nt) * 16 + lr;
            const unsigned short* wr = Wb1 + (size_t)col * CIN + kb + kh * 8;
            Frag bf;                       // B: lanes 0-15 K {0..7,16..23}, lanes 16-31 K {8..15,24..31}
            bf.q[0] = *(const uint4*)wr;
            bf.q[1] = *(const uint4*)(wr + 16);
            if (kb + 32 < CIN) __builtin_prefetch(wr + 32, 0, 1);
            acc[nt] = wmma_bf16(af, bf, acc[nt]);
        }
        __syncthreads();                               // single barrier per k-step
        cur ^= 1;
    }

    // epilogue: bias + ReLU -> LDS
    #pragma unroll
    for (int nt = 0; nt < 5; ++nt) {
        int col = (nw * 5 + nt) * 16 + lr;
        float bias = (col < NPROJ) ? b1[col] : ba1[col - NPROJ];
        #pragma unroll
        for (int v = 0; v < 8; ++v) {
            int m = mw * 16 + kh * 8 + v;
            float x = acc[nt][v] + bias;
            h1ls[m * NTOT + col] = f2bf(fmaxf(x, 0.f));
        }
    }
    if (tid < 32) normLDS[tid] = 0.f;
    __syncthreads();

    // pixel attention head: 64-dot + sigmoid, weight masked by slot
    if (tid < 32) {
        float a = ba2[0];
        for (int j = 0; j < NATT; ++j)
            a += bf2f(h1ls[tid * NTOT + NPROJ + j]) * Wa2[j];
        float w = 1.f / (1.f + __expf(-a));
        int sg = s0 + tid;
        float wv = (sg < cnts) ? w : 0.f;
        wsel[b * MAXS + sg] = wv;
        if (sg < cnts) atomicAdd(&sumw[b], wv);
    }

    // ---- GEMM2: h1(32 x 256) @ W2^T -> proj(32 x 256), then L2 normalize
    v8f acc2[4] = {};
    for (int kb = 0; kb < NPROJ; kb += 32) {
        Frag af;
        const unsigned short* ar = &h1ls[(mw * 16 + lr) * NTOT + kb + kh * 8];
        af.q[0] = *(const uint4*)ar;
        af.q[1] = *(const uint4*)(ar + 16);
        #pragma unroll
        for (int nt = 0; nt < 4; ++nt) {
            int col = (nw * 4 + nt) * 16 + lr;
            const unsigned short* wr = Wb2 + (size_t)col * NPROJ + kb + kh * 8;
            Frag bf;
            bf.q[0] = *(const uint4*)wr;
            bf.q[1] = *(const uint4*)(wr + 16);
            acc2[nt] = wmma_bf16(af, bf, acc2[nt]);
        }
    }
    #pragma unroll
    for (int nt = 0; nt < 4; ++nt) {
        int col = (nw * 4 + nt) * 16 + lr;
        #pragma unroll
        for (int v = 0; v < 8; ++v) {
            int m = mw * 16 + kh * 8 + v;
            float x = acc2[nt][v] + b2[col];
            atomicAdd(&normLDS[m], x * x);
        }
    }
    __syncthreads();
    #pragma unroll
    for (int nt = 0; nt < 4; ++nt) {
        int col = (nw * 4 + nt) * 16 + lr;
        #pragma unroll
        for (int v = 0; v < 8; ++v) {
            int m = mw * 16 + kh * 8 + v;
            float x = acc2[nt][v] + b2[col];
            float inv = 1.f / fmaxf(sqrtf(normLDS[m]), 1e-12f);
            fsn[((size_t)b * MAXS + s0 + m) * NPROJ + col] = f2bf(x * inv);
        }
    }
}

// --------------------------- K3: sim = fs @ fs^T (WMMA) fused with loss sums
// grid = BATCH*16 (one 16-row strip each); 128 threads = 4 waves (N split)
__global__ __launch_bounds__(128) void k_loss(
    const unsigned short* __restrict__ fsn,
    const float* __restrict__ wsel, const int* __restrict__ sel_cnt,
    float* numacc, int* hasflag) {

    __shared__ float wl[MAXS];
    __shared__ float d16[16], p16[16], q16[16];
    __shared__ int cnts;

    const int bx = blockIdx.x;
    const int b  = bx >> 4;
    const int ib = (bx & 15) * 16;            // row strip base
    const int tid = threadIdx.x;
    const int nw = tid >> 5, lane = tid & 31;
    const int lr = lane & 15, kh = lane >> 4;

    wl[tid]       = wsel[b * MAXS + tid];
    wl[tid + 128] = wsel[b * MAXS + tid + 128];
    if (tid < 16) { d16[tid] = 0.f; p16[tid] = 0.f; q16[tid] = 0.f; }
    if (tid == 0) cnts = sel_cnt[b];
    __syncthreads();

    v8f acc[4] = {};
    for (int kb = 0; kb < NPROJ; kb += 32) {
        Frag af;
        const unsigned short* ar =
            fsn + ((size_t)b * MAXS + ib + lr) * NPROJ + kb + kh * 8;
        af.q[0] = *(const uint4*)ar;
        af.q[1] = *(const uint4*)(ar + 16);
        #pragma unroll
        for (int nt = 0; nt < 4; ++nt) {
            int j0 = (nw * 4 + nt) * 16 + lr;   // B column n == fs row n
            const unsigned short* br =
                fsn + ((size_t)b * MAXS + j0) * NPROJ + kb + kh * 8;
            Frag bf;
            bf.q[0] = *(const uint4*)br;
            bf.q[1] = *(const uint4*)(br + 16);
            acc[nt] = wmma_bf16(af, bf, acc[nt]);
        }
    }

    const float invT = 10.0f;                 // 1/TEMP
    float dL[8] = {}, pL[8] = {}, qL[8] = {};
    #pragma unroll
    for (int nt = 0; nt < 4; ++nt) {
        int j = (nw * 4 + nt) * 16 + lr;
        bool sj = j < cnts;
        float wj = sj ? wl[j] : 0.f;
        #pragma unroll
        for (int v = 0; v < 8; ++v) {
            int i = ib + kh * 8 + v;
            float s = acc[nt][v];
            float e = sj ? __expf(s * invT) : 0.f;   // denom term (incl diag)
            bool pos = (s > 0.7f) && (i != j) && (i < cnts) && sj;
            float pw = pos ? wj : 0.f;
            dL[v] += e;
            pL[v] += pw;
            qL[v] += pw * s * invT;
        }
    }
    #pragma unroll
    for (int v = 0; v < 8; ++v) {
        atomicAdd(&d16[kh * 8 + v], dL[v]);
        atomicAdd(&p16[kh * 8 + v], pL[v]);
        atomicAdd(&q16[kh * 8 + v], qL[v]);
    }
    __syncthreads();

    if (tid < 16) {
        int i = ib + tid;
        if (i < cnts && p16[tid] > 0.f) {
            float den  = d16[tid];
            float logd = __logf(den > 0.f ? den : 1.f);
            float contrib = wl[i] * (q16[tid] - p16[tid] * logd);
            atomicAdd(&numacc[b], contrib);
            atomicOr(&hasflag[b], 1);
        }
    }
}

// ---------------------------------------------------------------- K4: final
__global__ void k_final(const float* numacc, const float* sumw,
                        const int* hasflag, float* out) {
    if (threadIdx.x == 0 && blockIdx.x == 0) {
        float total = 0.f; int n = 0;
        for (int b = 0; b < BATCH; ++b) {
            if (hasflag[b]) {
                total += -(0.1f / 0.07f) * numacc[b] / sumw[b];
                ++n;
            }
        }
        out[0] = (n > 0) ? total / (float)n * 0.1f : 0.f;
    }
}

extern "C" void kernel_launch(void* const* d_in, const int* in_sizes, int n_in,
                              void* d_out, int out_size, void* d_ws, size_t ws_size,
                              hipStream_t stream) {
    const float* feats  = (const float*)d_in[0];
    const int*   labels = (const int*)d_in[1];
    const float* W1  = (const float*)d_in[2];
    const float* b1  = (const float*)d_in[3];
    const float* W2  = (const float*)d_in[4];
    const float* b2  = (const float*)d_in[5];
    const float* Wa1 = (const float*)d_in[6];
    const float* ba1 = (const float*)d_in[7];
    const float* Wa2 = (const float*)d_in[8];
    const float* ba2 = (const float*)d_in[9];
    float* out = (float*)d_out;

    char* base = (char*)d_ws;                 // offsets kept 256B aligned
    int*            sel_idx = (int*)(base);                        // 8 KB
    int*            sel_cnt = (int*)(base + 8448);
    float*          wsel    = (float*)(base + 8704);               // 8 KB
    float*          sumw    = (float*)(base + 16896);
    float*          numacc  = (float*)(base + 17152);
    int*            hasflag = (int*)(base + 17408);
    unsigned short* fsn     = (unsigned short*)(base + 17920);     // 1 MB
    unsigned short* Wb1     = (unsigned short*)(base + 17920 + 1048576);   // 1.25 MB
    unsigned short* Wb2     = (unsigned short*)(base + 17920 + 1048576 + 1310720); // 128 KB

    k_select<<<BATCH, 32, 0, stream>>>(labels, sel_idx, sel_cnt,
                                       sumw, numacc, hasflag);
    k_cvtw<<<(NTOT * CIN + NPROJ * NPROJ + 255) / 256, 256, 0, stream>>>(
        W1, Wa1, W2, Wb1, Wb2);
    k_proj<<<BATCH * 8, 256, 0, stream>>>(feats, Wb1, b1, Wb2, b2, ba1,
                                          Wa2, ba2, sel_idx, sel_cnt,
                                          wsel, sumw, fsn);
    k_loss<<<BATCH * 16, 128, 0, stream>>>(fsn, wsel, sel_cnt, numacc, hasflag);
    k_final<<<1, 1, 0, stream>>>(numacc, sumw, hasflag, out);
}
